// JunmaiLayer_12575664243112
// MI455X (gfx1250) — matrix operand order repaired and checked by reading, not compiled, not run on hardware
//
#include <hip/hip_runtime.h>
#include <cstdint>

// N=512, R=16, H=16, OUT=1
#define NRH 131072  // N*R*H

typedef _Float16 half_t;
typedef __attribute__((ext_vector_type(16))) _Float16 v16h;
typedef __attribute__((ext_vector_type(8)))  _Float16 h8;
typedef __attribute__((ext_vector_type(8)))  float v8f;

union AFrag { v16h v; h8 h[2]; half_t e[16]; };

__device__ __forceinline__ v8f wmma_f16(v16h a, v16h b, v8f c) {
  // D = A(16x32 f16) * B(32x16 f16) + C(16x16 f32)
  return __builtin_amdgcn_wmma_f32_16x16x32_f16(false, a, false, b, (short)0, c,
                                                false, false);
}

__device__ __forceinline__ float fast_silu(float v) {
  return v * __builtin_amdgcn_rcpf(1.0f + __expf(-v));
}

// ---------------------------------------------------------------------------
// K0: one-time swizzle of K/Q weights (2 x 8192 x 16 fp32) into f16
// WMMA-B-fragment-native layout: wswz[w][kc][lane] = v16h. 512 KB, L2-resident.
// ---------------------------------------------------------------------------
__global__ __launch_bounds__(32) void k0_swizzle_w(
    const float* __restrict__ W, half_t* __restrict__ wswz)
{
  const int lane = threadIdx.x;
  const int kc = blockIdx.x;              // 0..255 K-chunks of 32
  const int m = lane & 15;
  const int kb = (lane >> 4) * 16;
  AFrag BK, BQ;
#pragma unroll
  for (int e = 0; e < 16; ++e) {
    int kk = kc*32 + (e & 8)*2 + (kb >> 1) + (e & 7);
    BK.e[e] = (half_t)W[(size_t)kk*16 + m];
    BQ.e[e] = (half_t)W[(size_t)NRH + (size_t)kk*16 + m];
  }
  v16h* p = (v16h*)wswz;
  p[(size_t)kc*32 + lane]         = BK.v;
  p[(size_t)(256 + kc)*32 + lane] = BQ.v;
}

// ---------------------------------------------------------------------------
// K1: compute basis on the fly (register-resident, WMMA A-layout), run the
// phase-1 GEMMs  bk/bq[i,c,b] = sum_{j,r} basis[i,j,r,b] * {K,Q}[j,r,c],
// plus sbasis[i,r,b] = sum_j basis via a constant one-hot B matrix (WMMA),
// and store f16 basis fragments for phase 2.
// grid (32 i-tiles, 16 j-slabs), 1 wave per block. 9 WMMAs per K-chunk.
// ---------------------------------------------------------------------------
__global__ __launch_bounds__(32) void k1_basis_phase1(
    const float* __restrict__ x, const half_t* __restrict__ wswz,
    half_t* __restrict__ basis_h, float* __restrict__ part,
    float* __restrict__ psb)
{
  const int lane = threadIdx.x;
  const int it = blockIdx.x;     // i-tile (16 rows)
  const int js = blockIdx.y;     // j-slab (32 j's)
  const int m  = lane & 15;
  const int hi = lane >> 4;
  const int rb = hi * 8;         // r-base this lane covers in A fragment
  const int kb = hi * 16;        // kb>>1 = 8*hi is the K-base this lane covers in B fragment
  const int i  = it * 16 + m;

  const float xi0 = x[i*3+0], xi1 = x[i*3+1], xi2 = x[i*3+2];

  const float startv = 0.00673794700f;                 // exp(-5)
  const float mstep  = (1.0f - startv) / 15.0f;        // linspace step
  const float beta   = 1.0f / (0.1241577566f * 0.1241577566f);

  const v16h* wf = (const v16h*)wswz;

  // constant one-hot B: B[(j,r'),n] = (r' == n); two stacked 16x16 identities
  AFrag BI;
#pragma unroll
  for (int e = 0; e < 16; ++e)
    BI.e[e] = ((((e & 8)*2 + (kb >> 1) + (e & 7)) & 15) == m) ? (half_t)1.0f : (half_t)0.0f;

  v8f aK0 = {}, aK1 = {}, aK2 = {}, aQ0 = {}, aQ1 = {}, aQ2 = {};
  v8f aS0 = {}, aS1 = {}, aS2 = {};

  for (int ch = 0; ch < 16; ++ch) {            // 16 chunks of K=32 (2 j's)
    const int j0 = js*32 + ch*2;
    AFrag A0, A1, A2;
#pragma unroll
    for (int jj = 0; jj < 2; ++jj) {
      const int j = j0 + jj;
      float dx = xi0 - x[j*3+0];
      float dy = xi1 - x[j*3+1];
      float dz = xi2 - x[j*3+2];
      float nsq = dx*dx + dy*dy + dz*dz + 1e-6f;
      float d   = __builtin_amdgcn_sqrtf(nsq);   // raw v_sqrt_f32 (TRANS)
      float inv = __builtin_amdgcn_rcpf(nsq);    // raw v_rcp_f32 (TRANS)
      float ex  = __expf(-d);
      float cut = (d < 5.0f) ? 0.5f * (__cosf(d * 0.62831853f) + 1.0f) : 0.0f;
      float bx = dx*inv, by = dy*inv, bz = dz*inv;
#pragma unroll
      for (int rr = 0; rr < 8; ++rr) {
        float mean = startv + mstep * (float)(rb + rr);
        float t = ex - mean;
        float sm = cut * __expf(-beta * t * t);
        int e = jj*8 + rr;
        A0.e[e] = (half_t)(sm*bx);
        A1.e[e] = (half_t)(sm*by);
        A2.e[e] = (half_t)(sm*bz);
      }
    }
    // pre-swizzled B fragments: single v16h load per matrix
    const size_t chg = (size_t)js*16 + ch;
    v16h BK = wf[chg*32 + lane];
    v16h BQ = wf[(256 + chg)*32 + lane];

    aK0 = wmma_f16(A0.v, BK, aK0);
    aK1 = wmma_f16(A1.v, BK, aK1);
    aK2 = wmma_f16(A2.v, BK, aK2);
    aQ0 = wmma_f16(A0.v, BQ, aQ0);
    aQ1 = wmma_f16(A1.v, BQ, aQ1);
    aQ2 = wmma_f16(A2.v, BQ, aQ2);
    aS0 = wmma_f16(A0.v, BI.v, aS0);
    aS1 = wmma_f16(A1.v, BI.v, aS1);
    aS2 = wmma_f16(A2.v, BI.v, aS2);

    // store basis fragments in WMMA-native layout for phase 2 reuse
    v16h* bp = (v16h*)(basis_h + (((size_t)it*256 + chg)*3)*512);
    bp[0*32 + lane] = A0.v;
    bp[1*32 + lane] = A1.v;
    bp[2*32 + lane] = A2.v;
  }

  // write partials: part[js][wb][row][c]; C-layout row = v + hi*8, col = m
  const v8f accs[6] = {aK0, aK1, aK2, aQ0, aQ1, aQ2};
#pragma unroll
  for (int wb = 0; wb < 6; ++wb)
#pragma unroll
    for (int v = 0; v < 8; ++v) {
      int row = it*16 + v + hi*8;
      part[(((size_t)js*6 + wb)*512 + row)*16 + m] = accs[wb][v];
    }
  // sbasis partials: psb[js][b][row][r] (C-layout: col = r = m)
  const v8f sacc[3] = {aS0, aS1, aS2};
#pragma unroll
  for (int b = 0; b < 3; ++b)
#pragma unroll
    for (int v = 0; v < 8; ++v) {
      int row = it*16 + v + hi*8;
      psb[(((size_t)js*3 + b)*512 + row)*16 + m] = sacc[b][v];
    }
  (void)i; (void)rb;
}

// ---------------------------------------------------------------------------
// K2: reduce partials -> bk1/bq1/sbasis, att1 = bk.bq, MLP1 (WMMA), write
// att_full (N x 1024 fp32) + pre-swizzled f16 B-fragments of the right half
// (cols 512..1024) for K3. 32 blocks, 1 wave each.
// ---------------------------------------------------------------------------
__global__ __launch_bounds__(32) void k2_mlp1(
    const float* __restrict__ part, const float* __restrict__ psb,
    const float* __restrict__ W1, const float* __restrict__ b1,
    const float* __restrict__ W2,
    float* __restrict__ attf, float* __restrict__ sb,
    half_t* __restrict__ frag2)
{
  __shared__ __align__(16) float  bkq[2][3][16][16];  // [w][b][m][c]
  __shared__ __align__(16) half_t a1[16][256];
  __shared__ __align__(16) half_t tT[16][32];
  __shared__ __align__(16) half_t rt[16][512];        // right cols as f16

  const int lane = threadIdx.x;
  const int it = blockIdx.x;
  const int m = lane & 15;
  const int hi = lane >> 4;
  const int kb = hi * 16;
  const int kb8 = hi * 8;

  for (int idx = lane; idx < 1536; idx += 32) {
    int wb = idx >> 8;
    int mm = (idx >> 4) & 15;
    int c  = idx & 15;
    int row = it*16 + mm;
    float s = 0.f;
    for (int jsl = 0; jsl < 16; ++jsl)
      s += part[(((size_t)jsl*6 + wb)*512 + row)*16 + c];
    ((float*)bkq)[idx] = s;
  }
  for (int idx = lane; idx < 768; idx += 32) {
    int mm = idx / 48, rb3 = idx % 48;
    int r = rb3 / 3, b = rb3 % 3;
    int row = it*16 + mm;
    float s = 0.f;
    for (int jsl = 0; jsl < 16; ++jsl)
      s += psb[(((size_t)jsl*3 + b)*512 + row)*16 + r];
    sb[(size_t)row*48 + rb3] = s;
  }
  __syncthreads();
  for (int idx = lane; idx < 4096; idx += 32) {
    int mm = idx >> 8, hg = idx & 255, h = hg >> 4, g = hg & 15;
    float s = bkq[0][0][mm][h]*bkq[1][0][mm][g]
            + bkq[0][1][mm][h]*bkq[1][1][mm][g]
            + bkq[0][2][mm][h]*bkq[1][2][mm][g];
    a1[mm][hg] = (half_t)s;
  }
  __syncthreads();

  // GEMM1: (16x256) @ W1^T (256x16), K in 8 chunks of 32
  v8f acc = {};
  for (int chnk = 0; chnk < 8; ++chnk) {
    const int k0 = chnk * 32;
    AFrag A, B;
    A.h[0] = *(const h8*)&a1[m][k0 + kb8];
    A.h[1] = *(const h8*)&a1[m][k0 + 16 + kb8];
#pragma unroll
    for (int e = 0; e < 16; ++e)
      B.e[e] = (half_t)W1[(size_t)m*256 + k0 + (e & 8)*2 + kb8 + (e & 7)];
    acc = wmma_f16(A.v, B.v, acc);
  }
  // bias + silu, transpose C-layout -> A-layout via LDS, zero-pad K 16..31
#pragma unroll
  for (int v = 0; v < 8; ++v) {
    float val = fast_silu(acc[v] + b1[m]);
    tT[v + kb8][m] = (half_t)val;
    tT[v + kb8][16 + m] = (half_t)0.f;
  }
  __syncthreads();
  AFrag tA;
  tA.h[0] = *(const h8*)&tT[m][kb8];
  tA.h[1] = *(const h8*)&tT[m][16 + kb8];

  // GEMM2: (16x16, K padded to 32) @ W2^T (16x1024) in 64 n-tiles
  for (int nt = 0; nt < 64; ++nt) {
    AFrag B;
#pragma unroll
    for (int e = 0; e < 16; ++e) {
      float wv = (e >= 8) ? 0.0f : W2[(size_t)(nt*16 + m)*16 + kb8 + e];
      B.e[e] = (half_t)wv;
    }
    v8f z = {};
    v8f o = wmma_f16(tA.v, B.v, z);
#pragma unroll
    for (int v = 0; v < 8; ++v) {
      attf[(size_t)(it*16 + v + kb8)*1024 + nt*16 + m] = o[v];
      if (nt >= 32)                          // keep right half as f16 in LDS
        rt[v + kb8][(nt - 32)*16 + m] = (half_t)o[v];
    }
  }
  __syncthreads();

  // Build pre-swizzled B fragments for K3 from this block's 16 rows
  // (they cover global K-chunks kc = it*8 .. it*8+8).
  v16h* f2 = (v16h*)frag2;
#pragma unroll
  for (int kcl = 0; kcl < 8; ++kcl) {
    AFrag BR, BS;
#pragma unroll
    for (int e = 0; e < 16; ++e) {
      int kk = kcl*32 + (e & 8)*2 + (kb >> 1) + (e & 7);
      int jl = kk >> 4, r = kk & 15;
      BR.e[e] = rt[jl][r*16 + m];          // cols 512..768
      BS.e[e] = rt[jl][256 + r*16 + m];    // cols 768..1024
    }
    size_t kc = (size_t)it*8 + kcl;
    f2[kc*32 + lane]         = BR.v;
    f2[(256 + kc)*32 + lane] = BS.v;
  }
}

// ---------------------------------------------------------------------------
// K3: phase-2 "right" GEMM: stored basis fragments x pre-swizzled att
// fragments. grid (32, 16), 1 wave per block; pure fragment traffic.
// ---------------------------------------------------------------------------
__global__ __launch_bounds__(32) void k3_phase2(
    const half_t* __restrict__ basis_h, const half_t* __restrict__ frag2,
    float* __restrict__ part)
{
  const int lane = threadIdx.x;
  const int it = blockIdx.x;
  const int js = blockIdx.y;
  const int m = lane & 15;
  const int hi = lane >> 4;
  const v16h* f2 = (const v16h*)frag2;

  v8f aK0={},aK1={},aK2={},aQ0={},aQ1={},aQ2={};
  for (int ch = 0; ch < 16; ++ch) {
    const size_t chg = (size_t)js*16 + ch;
    const v16h* bp = (const v16h*)(basis_h + (((size_t)it*256 + chg)*3)*512);
    v16h A0 = bp[0*32+lane], A1 = bp[1*32+lane], A2 = bp[2*32+lane];
    v16h BR = f2[chg*32 + lane];
    v16h BS = f2[(256 + chg)*32 + lane];
    aK0 = wmma_f16(A0, BR, aK0);
    aK1 = wmma_f16(A1, BR, aK1);
    aK2 = wmma_f16(A2, BR, aK2);
    aQ0 = wmma_f16(A0, BS, aQ0);
    aQ1 = wmma_f16(A1, BS, aQ1);
    aQ2 = wmma_f16(A2, BS, aQ2);
  }
  const v8f accs[6] = {aK0,aK1,aK2,aQ0,aQ1,aQ2};
#pragma unroll
  for (int wb = 0; wb < 6; ++wb)
#pragma unroll
    for (int v = 0; v < 8; ++v) {
      int row = it*16 + v + hi*8;
      part[(((size_t)js*6 + wb)*512 + row)*16 + m] = accs[wb][v];
    }
}

// ---------------------------------------------------------------------------
// K4: reduce phase-2 partials, add rank-R "left" correction via sbasis,
// att2 = bk2.bq2, MLP2 (WMMA) + final dot with W4. 32 blocks, 1 wave each.
// ---------------------------------------------------------------------------
__global__ __launch_bounds__(32) void k4_final(
    const float* __restrict__ part, const float* __restrict__ sb,
    const float* __restrict__ attf,
    const float* __restrict__ W3, const float* __restrict__ b3,
    const float* __restrict__ W4, const float* __restrict__ b4,
    float* __restrict__ out)
{
  __shared__ __align__(16) float  bkq[2][3][16][16];
  __shared__ __align__(16) half_t a2[16][256];
  __shared__ __align__(16) float  tf[16][16];

  const int lane = threadIdx.x;
  const int it = blockIdx.x;
  const int m = lane & 15;
  const int hi = lane >> 4;
  const int kb = hi*16, kb8 = hi*8;

  for (int idx = lane; idx < 1536; idx += 32) {
    int wb = idx >> 8;
    int w = wb / 3, b = wb % 3;
    int mm = (idx >> 4) & 15;
    int c = idx & 15;
    int row = it*16 + mm;
    float s = 0.f;
    for (int jsl = 0; jsl < 16; ++jsl)
      s += part[(((size_t)jsl*6 + wb)*512 + row)*16 + c];
    // left contribution: sum_r sbasis[row,r,b] * att[row, off + r*16 + c]
    int off = w ? 256 : 0;
    for (int r = 0; r < 16; ++r)
      s += sb[(size_t)row*48 + r*3 + b] * attf[(size_t)row*1024 + off + r*16 + c];
    ((float*)bkq)[idx] = s;
  }
  __syncthreads();
  for (int idx = lane; idx < 4096; idx += 32) {
    int mm = idx >> 8, hg = idx & 255, h = hg >> 4, g = hg & 15;
    float s = bkq[0][0][mm][h]*bkq[1][0][mm][g]
            + bkq[0][1][mm][h]*bkq[1][1][mm][g]
            + bkq[0][2][mm][h]*bkq[1][2][mm][g];
    a2[mm][hg] = (half_t)s;
  }
  __syncthreads();
  v8f acc = {};
  for (int chnk = 0; chnk < 8; ++chnk) {
    const int k0 = chnk*32;
    AFrag A, B;
    A.h[0] = *(const h8*)&a2[m][k0 + kb8];
    A.h[1] = *(const h8*)&a2[m][k0 + 16 + kb8];
#pragma unroll
    for (int e = 0; e < 16; ++e)
      B.e[e] = (half_t)W3[(size_t)m*256 + k0 + (e & 8)*2 + kb8 + (e & 7)];
    acc = wmma_f16(A.v, B.v, acc);
  }
#pragma unroll
  for (int v = 0; v < 8; ++v)
    tf[v + kb8][m] = fast_silu(acc[v] + b3[m]);
  __syncthreads();
  if (lane < 16) {
    float s = b4[0];
#pragma unroll
    for (int c = 0; c < 16; ++c)
      s += tf[lane][c] * W4[c];
    out[it*16 + lane] = s;
  }
}

// ---------------------------------------------------------------------------
extern "C" void kernel_launch(void* const* d_in, const int* in_sizes, int n_in,
                              void* d_out, int out_size, void* d_ws, size_t ws_size,
                              hipStream_t stream) {
  const float* x  = (const float*)d_in[0];
  const float* W  = (const float*)d_in[1];
  const float* W1 = (const float*)d_in[2];
  const float* b1 = (const float*)d_in[3];
  const float* W2 = (const float*)d_in[4];
  const float* W3 = (const float*)d_in[5];
  const float* b3 = (const float*)d_in[6];
  const float* W4 = (const float*)d_in[7];
  const float* b4 = (const float*)d_in[8];
  float* out = (float*)d_out;

  char* ws = (char*)d_ws;
  half_t* basis_h = (half_t*)(ws);              // 25,165,824 B: f16 basis frags
  float*  part    = (float*)(ws + 25165824);    //  3,145,728 B: GEMM partials (reused by K3)
  float*  psb     = (float*)(ws + 28311552);    //  1,572,864 B: sbasis partials
  float*  attf    = (float*)(ws + 29884416);    //  2,097,152 B: att_full (N x 1024)
  float*  sb      = (float*)(ws + 31981568);    //     98,304 B: reduced sbasis
  half_t* wswz    = (half_t*)(ws + 32079872);   //    524,288 B: f16 weight frags
  half_t* frag2   = (half_t*)(ws + 32604160);   //    524,288 B: f16 att frags for K3

  k0_swizzle_w   <<<256,         32, 0, stream>>>(W, wswz);
  k1_basis_phase1<<<dim3(32,16), 32, 0, stream>>>(x, wswz, basis_h, part, psb);
  k2_mlp1        <<<32,          32, 0, stream>>>(part, psb, W1, b1, W2, attf, sb, frag2);
  k3_phase2      <<<dim3(32,16), 32, 0, stream>>>(basis_h, frag2, part);
  k4_final       <<<32,          32, 0, stream>>>(part, sb, attf, W3, b3, W4, b4, out);

  (void)in_sizes; (void)n_in; (void)out_size; (void)ws_size;
}
